// DeepseekV2Attention_31645319037702
// MI455X (gfx1250) — matrix operand order repaired and checked by reading, not compiled, not run on hardware
//
#include <hip/hip_runtime.h>
#include <math.h>

// ---------------------------------------------------------------------------
// DeepSeek-V2 MLA prefill for MI455X (gfx1250, wave32, WMMA bf16).
// Round 2: eliminate GEMM accumulator spills (launch_bounds + compile-time K
// so sub-tile offsets fold into global_load immediate offsets), hoist all
// loop-invariant fragments/pointers in attention, LDS-tiled transpose.
// ---------------------------------------------------------------------------

#define T_SEQ   2048
#define HEADS   16
#define D_NOPE  128
#define D_ROPE  64
#define D_V     128
#define QK_DIM  192           // D_NOPE + D_ROPE
#define KV_RANK 512
#define Q_RANK  1536
#define HIDDEN  2048
#define Q_OUT   (HEADS * QK_DIM)          // 3072
#define KV_OUT  (HEADS * (D_NOPE + D_V))  // 4096
#define LAT_DIM (KV_RANK + D_ROPE)        // 576
#define SCALE_F 0.072168783648703220563f  // 192^-0.5
#define NEG_INF (-3.0e38f)

typedef __attribute__((ext_vector_type(16))) __bf16 v16bf;
typedef __attribute__((ext_vector_type(8)))  __bf16 v8bf;
typedef __attribute__((ext_vector_type(8)))  float  v8f;

static __device__ __forceinline__ __bf16 f2bf(float f) {
  union { float f; unsigned u; } in; in.f = f;
  unsigned u = in.u;
  unsigned r = u + 0x7FFFu + ((u >> 16) & 1u);     // round-to-nearest-even
  union { unsigned short s; __bf16 b; } out;
  out.s = (unsigned short)(r >> 16);
  return out.b;
}

static __device__ __forceinline__ v16bf ldpair(const __bf16* p0, const __bf16* p1) {
  v8bf lo = *(const v8bf*)p0;
  v8bf hi = *(const v8bf*)p1;
  return __builtin_shufflevector(lo, hi, 0,1,2,3,4,5,6,7,8,9,10,11,12,13,14,15);
}

static __device__ __forceinline__ v8f wmma_bf16(v16bf a, v16bf b, v8f c) {
  // D(16x16,f32) = A(16x32,bf16) x B(32x16,bf16) + C
  return __builtin_amdgcn_wmma_f32_16x16x32_bf16(
      /*neg_a=*/false, a, /*neg_b=*/false, b,
      /*c_mod=*/(short)0, c, /*reuse_a=*/false, /*reuse_b=*/false);
}

// --------------------------- elementwise f32 -> bf16 -----------------------
__global__ void k_cvt_bf16(const float* __restrict__ in, __bf16* __restrict__ out,
                           long n) {
  for (long i = blockIdx.x * (long)blockDim.x + threadIdx.x; i < n;
       i += (long)gridDim.x * blockDim.x)
    out[i] = f2bf(in[i]);
}

// -------- transpose KxN f32 -> NxK bf16 (LDS 32x32 tiles, coalesced) -------
__global__ void k_transpose_bf16(const float* __restrict__ in, __bf16* __restrict__ out,
                                 int K, int N) {
  __shared__ float tile[32][33];
  const int kb = blockIdx.x * 32, nb = blockIdx.y * 32;
#pragma unroll
  for (int r = threadIdx.y; r < 32; r += 8) {
    int k = kb + r, n = nb + threadIdx.x;
    if (k < K && n < N) tile[r][threadIdx.x] = in[(long)k * N + n];
  }
  __syncthreads();
#pragma unroll
  for (int r = threadIdx.y; r < 32; r += 8) {
    int n = nb + r, k = kb + threadIdx.x;
    if (n < N && k < K) out[(long)n * K + k] = f2bf(tile[threadIdx.x][r]);
  }
}

// ------------------------------- RMSNorm -----------------------------------
__global__ void k_rmsnorm_bf16(const float* __restrict__ in, const float* __restrict__ w,
                               __bf16* __restrict__ out, int cols, int in_stride) {
  int row = blockIdx.x;
  const float* x = in + (long)row * in_stride;
  __shared__ float sred[256];
  float ss = 0.f;
  for (int c = threadIdx.x; c < cols; c += blockDim.x) { float v = x[c]; ss += v * v; }
  sred[threadIdx.x] = ss;
  __syncthreads();
  for (int s = 128; s > 0; s >>= 1) {
    if ((int)threadIdx.x < s) sred[threadIdx.x] += sred[threadIdx.x + s];
    __syncthreads();
  }
  float inv = rsqrtf(sred[0] / (float)cols + 1e-6f);
  for (int c = threadIdx.x; c < cols; c += blockDim.x)
    out[(long)row * cols + c] = f2bf(x[c] * inv * w[c]);
}

// ------------------------------ generic GEMM -------------------------------
// C[M,N](f32) = A[M,K](bf16, row-major) * Bt[N,K](bf16, row-major = B^T)
// One wave computes a 64x64 tile (4x4 wmma accumulators = 128 VGPRs).
// K is a compile-time constant: sub-tile strides (i*16*K elements) fold into
// the 24-bit global_load immediate offset, so only 2 pointers live in loop.
template <int K>
__global__ __launch_bounds__(256, 1) void k_gemm_bf16(
    const __bf16* __restrict__ A, const __bf16* __restrict__ Bt,
    float* __restrict__ C, int M, int N) {
  const int lane = threadIdx.x & 31;
  const int wave = blockIdx.x * (blockDim.x >> 5) + (threadIdx.x >> 5);
  const int tilesN = N >> 6;
  const int tilesTotal = (M >> 6) * tilesN;
  if (wave >= tilesTotal) return;
  const int tm = (wave / tilesN) << 6;
  const int tn = (wave % tilesN) << 6;
  const int l16 = lane & 15, half = lane >> 4;

  const v8f vzero = {0.f,0.f,0.f,0.f,0.f,0.f,0.f,0.f};
  v8f acc[4][4];
#pragma unroll
  for (int i = 0; i < 4; ++i)
#pragma unroll
    for (int j = 0; j < 4; ++j) acc[i][j] = vzero;

  // per-lane base pointers (incremented by 32 elements / 64 B each K step)
  const __bf16* ap = A  + (long)(tm + l16) * K + half * 8;   // A lane halves: +0 / +8
  const __bf16* bp = Bt + (long)(tn + l16) * K + half * 8;   // B lane halves: +0 / +8

  for (int k0 = 0; k0 < K; k0 += 32) {
    v16bf af[4], bfr[4];
#pragma unroll
    for (int i = 0; i < 4; ++i)
      af[i] = ldpair(ap + (long)i * 16 * K, ap + (long)i * 16 * K + 16);
#pragma unroll
    for (int j = 0; j < 4; ++j)
      bfr[j] = ldpair(bp + (long)j * 16 * K, bp + (long)j * 16 * K + 16);
    __builtin_prefetch(ap + 64, 0, 1);             // global_prefetch_b8
    __builtin_prefetch(bp + 64, 0, 1);
#pragma unroll
    for (int i = 0; i < 4; ++i)
#pragma unroll
      for (int j = 0; j < 4; ++j)
        acc[i][j] = wmma_bf16(af[i], bfr[j], acc[i][j]);
    ap += 32;
    bp += 32;
  }

#pragma unroll
  for (int i = 0; i < 4; ++i)
#pragma unroll
    for (int j = 0; j < 4; ++j) {
      int col = tn + 16 * j + l16;
#pragma unroll
      for (int r = 0; r < 8; ++r) {
        int row = tm + 16 * i + r + 8 * half;
        C[(long)row * N + col] = acc[i][j][r];
      }
    }
}

// --------------------- assemble qf / kf / v^T with RoPE --------------------
// qf[t][h][0:192]  = [q_nope, rope(q_pe)]           (bf16)
// kf[t][h][0:192]  = [k_nope, rope(k_pe) bcast h]   (bf16)
// vt[h*128+d][t]   = kv[t][h*256 + 128 + d]         (bf16, keys contiguous)
__global__ void k_assemble(const float* __restrict__ q, const float* __restrict__ kv,
                           const float* __restrict__ latent, const int* __restrict__ pos,
                           __bf16* __restrict__ qf, __bf16* __restrict__ kf,
                           __bf16* __restrict__ vt) {
  int t = blockIdx.x;
  __shared__ float cs[32], sn[32];
  if (threadIdx.x < 32) {
    float p = (float)pos[t];
    float inv_freq = __expf(-logf(10000.f) * ((float)(2 * threadIdx.x) / 64.f));
    float f = p * inv_freq;
    cs[threadIdx.x] = __cosf(f);
    sn[threadIdx.x] = __sinf(f);
  }
  __syncthreads();

  const float* qrow = q + (long)t * Q_OUT;
  const float* kvrow = kv + (long)t * KV_OUT;
  const float* lrow = latent + (long)t * LAT_DIM;

  for (int i = threadIdx.x; i < HEADS * QK_DIM; i += blockDim.x) {
    int h = i / QK_DIM, d = i % QK_DIM;
    // qf
    float qv;
    if (d < D_NOPE) {
      qv = qrow[h * QK_DIM + d];
    } else {
      int j = d - D_NOPE, p2 = j >> 1;
      float x1 = qrow[h * QK_DIM + D_NOPE + 2 * p2];
      float x2 = qrow[h * QK_DIM + D_NOPE + 2 * p2 + 1];
      qv = (j & 1) ? (x2 * cs[p2] + x1 * sn[p2]) : (x1 * cs[p2] - x2 * sn[p2]);
    }
    qf[(long)t * Q_OUT + i] = f2bf(qv);
    // kf
    float kvv;
    if (d < D_NOPE) {
      kvv = kvrow[h * (D_NOPE + D_V) + d];
    } else {
      int j = d - D_NOPE, p2 = j >> 1;
      float x1 = lrow[KV_RANK + 2 * p2];
      float x2 = lrow[KV_RANK + 2 * p2 + 1];
      kvv = (j & 1) ? (x2 * cs[p2] + x1 * sn[p2]) : (x1 * cs[p2] - x2 * sn[p2]);
    }
    kf[(long)t * Q_OUT + i] = f2bf(kvv);
  }
  for (int i = threadIdx.x; i < HEADS * D_V; i += blockDim.x) {
    int h = i / D_V, d = i % D_V;
    vt[(long)(h * D_V + d) * T_SEQ + t] = f2bf(kvrow[h * (D_NOPE + D_V) + D_NOPE + d]);
  }
}

// ------------------------- flash attention (per wave) ----------------------
// Each wave: one head, 16 queries, streams keys in chunks of 32.
// St = K_tile(16x32feat as A) x Q^T(B)  -> D layout: rows=keys, cols=queries
//   => softmax stats are lane-local over 8 VGPRs + one shfl_xor(16).
// PV: A = V^T tiles (dv x keys), B = probs from LDS  -> O^T (dv x queries).
// Q fragments are key-loop invariant: preloaded once (bq[6], 48 VGPRs).
__global__ __launch_bounds__(256, 1) void k_attention(
    const __bf16* __restrict__ qf, const __bf16* __restrict__ kf,
    const __bf16* __restrict__ vt, __bf16* __restrict__ attn) {
  const int lane = threadIdx.x & 31;
  const int wid = threadIdx.x >> 5;
  const int task = blockIdx.x * 8 + wid;           // 2048 tasks = 128 qtiles x 16 heads
  if (task >= (T_SEQ / 16) * HEADS) return;
  const int h = task & (HEADS - 1);
  const int qbase = (task >> 4) * 16;
  const int l16 = lane & 15, half = lane >> 4;
  const int myq = qbase + l16;

  __shared__ __attribute__((aligned(16))) __bf16 plds_all[8][16 * 32];
  __bf16* plds = plds_all[wid];

  const v8f vzero = {0.f,0.f,0.f,0.f,0.f,0.f,0.f,0.f};
  v8f acc[8];                                      // O^T: 8 tiles of (16 dv x 16 q)
#pragma unroll
  for (int a = 0; a < 8; ++a) acc[a] = vzero;
  float m_run = NEG_INF, l_run = 0.f;

  // --- preload Q fragments (B operand), invariant over key chunks ---
  const __bf16* qp = qf + ((long)(qbase + l16) * HEADS + h) * QK_DIM + half * 8;
  v16bf bq[6];
#pragma unroll
  for (int fs = 0; fs < 6; ++fs)
    bq[fs] = ldpair(qp + fs * 32, qp + fs * 32 + 16);

  // --- streaming base pointers (compile-time tile offsets fold into insns) ---
  const __bf16* kp = kf + ((long)l16 * HEADS + h) * QK_DIM + half * 8;
  const __bf16* vp = vt + ((long)h * D_V + l16) * T_SEQ + half * 8;
  const long kstep = (long)32 * HEADS * QK_DIM;    // kf elements per 32-key chunk
  const long ktile = (long)16 * HEADS * QK_DIM;    // kf elements per 16-key sub-tile

  for (int kc = 0; kc * 32 <= qbase + 15; ++kc) {
    const int kbase = kc * 32;

    // ---- scores: St[kt] (16 keys x 16 queries), 6 K-feature steps of 32 ----
    v8f st[2] = {vzero, vzero};
#pragma unroll
    for (int fs = 0; fs < 6; ++fs) {
#pragma unroll
      for (int kt = 0; kt < 2; ++kt) {
        const __bf16* ap = kp + kt * ktile + fs * 32;
        v16bf ak = ldpair(ap, ap + 16);
        st[kt] = wmma_bf16(ak, bq[fs], st[kt]);
      }
    }

    // ---- scale + causal mask + per-query max ----
    float mx = m_run;
#pragma unroll
    for (int kt = 0; kt < 2; ++kt)
#pragma unroll
      for (int r = 0; r < 8; ++r) {
        int key = kbase + kt * 16 + r + 8 * half;
        float v = (key > myq) ? NEG_INF : st[kt][r] * SCALE_F;
        st[kt][r] = v;
        mx = fmaxf(mx, v);
      }
    mx = fmaxf(mx, __shfl_xor(mx, 16, 32));

    float alpha = __expf(m_run - mx);
    float sum = 0.f;
#pragma unroll
    for (int kt = 0; kt < 2; ++kt)
#pragma unroll
      for (int r = 0; r < 8; ++r) {
        float p = __expf(st[kt][r] - mx);
        st[kt][r] = p;
        sum += p;
      }
    sum += __shfl_xor(sum, 16, 32);
    l_run = l_run * alpha + sum;
    m_run = mx;
#pragma unroll
    for (int a = 0; a < 8; ++a)
#pragma unroll
      for (int e = 0; e < 8; ++e) acc[a][e] *= alpha;

    // ---- probs -> LDS (query-major, keys contiguous) ----
#pragma unroll
    for (int kt = 0; kt < 2; ++kt) {
      v8bf pb;
#pragma unroll
      for (int r = 0; r < 8; ++r) pb[r] = f2bf(st[kt][r]);
      *(v8bf*)(plds + l16 * 32 + kt * 16 + 8 * half) = pb;
    }
    asm volatile("s_wait_dscnt 0" ::: "memory");   // cross-lane LDS visibility

    // ---- B = probs (32 keys x 16 queries) from LDS ----
    v16bf bP = ldpair(plds + l16 * 32 + half * 8, plds + l16 * 32 + half * 8 + 16);

    // ---- PV: O^T += V^T x P  (a*16*T_SEQ folds into insn offset) ----
#pragma unroll
    for (int a = 0; a < 8; ++a) {
      const __bf16* vap = vp + (long)a * 16 * T_SEQ;
      v16bf av = ldpair(vap, vap + 16);
      acc[a] = wmma_bf16(av, bP, acc[a]);
    }

    kp += kstep;
    vp += 32;
  }

  // ---- epilogue: divide by row sum, store attn[t][h*128 + dv] as bf16 ----
  float invl = 1.f / l_run;
#pragma unroll
  for (int a = 0; a < 8; ++a) {
    v8bf ob;
#pragma unroll
    for (int r = 0; r < 8; ++r) ob[r] = f2bf(acc[a][r] * invl);
    *(v8bf*)(attn + (long)myq * (HEADS * D_V) + h * D_V + a * 16 + 8 * half) = ob;
  }
}

// ---------------------------------------------------------------------------
extern "C" void kernel_launch(void* const* d_in, const int* in_sizes, int n_in,
                              void* d_out, int out_size, void* d_ws, size_t ws_size,
                              hipStream_t stream) {
  const int*   positions = (const int*)d_in[0];
  const float* hidden    = (const float*)d_in[1];
  const float* q_a_w     = (const float*)d_in[2];
  const float* q_a_ln_w  = (const float*)d_in[3];
  const float* q_b_w     = (const float*)d_in[4];
  const float* kv_a_w    = (const float*)d_in[5];
  const float* kv_a_ln_w = (const float*)d_in[6];
  const float* kv_b_w    = (const float*)d_in[7];
  const float* o_w       = (const float*)d_in[8];
  float* out = (float*)d_out;

  // workspace carve-up (256B aligned); total ~166 MB
  char* base = (char*)d_ws;
  size_t off = 0;
  auto alloc = [&](size_t bytes) {
    size_t o = off;
    off = (off + bytes + 255) & ~(size_t)255;
    return o;
  };
  __bf16* hidden_bf = (__bf16*)(base + alloc((size_t)T_SEQ * HIDDEN * 2));
  __bf16* qaw_t  = (__bf16*)(base + alloc((size_t)Q_RANK * HIDDEN * 2));
  __bf16* qbw_t  = (__bf16*)(base + alloc((size_t)Q_OUT * Q_RANK * 2));
  __bf16* kvaw_t = (__bf16*)(base + alloc((size_t)LAT_DIM * HIDDEN * 2));
  __bf16* kvbw_t = (__bf16*)(base + alloc((size_t)KV_OUT * KV_RANK * 2));
  __bf16* ow_t   = (__bf16*)(base + alloc((size_t)HIDDEN * (HEADS * D_V) * 2));
  float*  q_a    = (float*)(base + alloc((size_t)T_SEQ * Q_RANK * 4));
  __bf16* q_a_n  = (__bf16*)(base + alloc((size_t)T_SEQ * Q_RANK * 2));
  float*  qbuf   = (float*)(base + alloc((size_t)T_SEQ * Q_OUT * 4));
  float*  latent = (float*)(base + alloc((size_t)T_SEQ * LAT_DIM * 4));
  __bf16* kv_a_n = (__bf16*)(base + alloc((size_t)T_SEQ * KV_RANK * 2));
  float*  kvbuf  = (float*)(base + alloc((size_t)T_SEQ * KV_OUT * 4));
  __bf16* qfb    = (__bf16*)(base + alloc((size_t)T_SEQ * Q_OUT * 2));
  __bf16* kfb    = (__bf16*)(base + alloc((size_t)T_SEQ * Q_OUT * 2));
  __bf16* vtb    = (__bf16*)(base + alloc((size_t)HEADS * D_V * T_SEQ * 2));
  __bf16* attnb  = (__bf16*)(base + alloc((size_t)T_SEQ * HEADS * D_V * 2));
  (void)ws_size; (void)n_in; (void)in_sizes; (void)out_size;

  // 1) convert activations / transpose+convert weights to bf16
  k_cvt_bf16<<<1024, 256, 0, stream>>>(hidden, hidden_bf, (long)T_SEQ * HIDDEN);
  {
    dim3 blk(32, 8);
    auto tgrid = [](int K, int N) { return dim3((K + 31) / 32, (N + 31) / 32); };
    k_transpose_bf16<<<tgrid(HIDDEN, Q_RANK), blk, 0, stream>>>(q_a_w, qaw_t, HIDDEN, Q_RANK);
    k_transpose_bf16<<<tgrid(Q_RANK, Q_OUT), blk, 0, stream>>>(q_b_w, qbw_t, Q_RANK, Q_OUT);
    k_transpose_bf16<<<tgrid(HIDDEN, LAT_DIM), blk, 0, stream>>>(kv_a_w, kvaw_t, HIDDEN, LAT_DIM);
    k_transpose_bf16<<<tgrid(KV_RANK, KV_OUT), blk, 0, stream>>>(kv_b_w, kvbw_t, KV_RANK, KV_OUT);
    k_transpose_bf16<<<tgrid(HEADS * D_V, HIDDEN), blk, 0, stream>>>(o_w, ow_t, HEADS * D_V, HIDDEN);
  }

#define GEMM(Kc, A, Bt, C, M, N)                                            \
  do {                                                                      \
    int tiles = ((M) / 64) * ((N) / 64);                                    \
    int blocks = (tiles + 7) / 8; /* 8 waves (64x64 tiles) per block */     \
    k_gemm_bf16<Kc><<<blocks, 256, 0, stream>>>((A), (Bt), (C), (M), (N));  \
  } while (0)

  // 2) q path
  GEMM(HIDDEN, hidden_bf, qaw_t, q_a, T_SEQ, Q_RANK);
  k_rmsnorm_bf16<<<T_SEQ, 256, 0, stream>>>(q_a, q_a_ln_w, q_a_n, Q_RANK, Q_RANK);
  GEMM(Q_RANK, q_a_n, qbw_t, qbuf, T_SEQ, Q_OUT);

  // 3) kv path
  GEMM(HIDDEN, hidden_bf, kvaw_t, latent, T_SEQ, LAT_DIM);
  k_rmsnorm_bf16<<<T_SEQ, 256, 0, stream>>>(latent, kv_a_ln_w, kv_a_n, KV_RANK, LAT_DIM);
  GEMM(KV_RANK, kv_a_n, kvbw_t, kvbuf, T_SEQ, KV_OUT);

  // 4) RoPE + layout assembly
  k_assemble<<<T_SEQ, 256, 0, stream>>>(qbuf, kvbuf, latent, positions, qfb, kfb, vtb);

  // 5) attention: 2048 wave-tasks (128 query tiles x 16 heads), 8 waves/block
  k_attention<<<256, 256, 0, stream>>>(qfb, kfb, vtb, attnb);

  // 6) output projection -> f32 d_out
  GEMM(HEADS * D_V, attnb, ow_t, out, T_SEQ, HIDDEN);
#undef GEMM
}
